// MutiHeadAttention_1013612282425
// MI455X (gfx1250) — matrix operand order repaired and checked by reading, not compiled, not run on hardware
//
#include <hip/hip_runtime.h>

typedef unsigned short u16;
typedef __attribute__((ext_vector_type(16))) __bf16 v16bf;
typedef __attribute__((ext_vector_type(8)))  float  v8f;

union FragU { v16bf bf; uint4 q[2]; };

#define B_   4
#define L_   2048
#define NH   8

__device__ __forceinline__ u16 f2bf(float f) {
  unsigned u = __float_as_uint(f);
  u += 0x7fffu + ((u >> 16) & 1u);       // round-to-nearest-even
  return (u16)(u >> 16);
}

// Async global->LDS copy (CDNA5, ASYNCcnt-tracked).  lds_off is the LDS byte
// address = low 32 bits of the generic pointer (flat LDS aperture keeps the
// LDS offset in addr[31:0]).
__device__ __forceinline__ void async_cp16(unsigned lds_off,
                                           const u16* __restrict__ g) {
  asm volatile("global_load_async_to_lds_b128 %0, %1, off"
               :: "v"(lds_off), "v"(g) : "memory");
}
__device__ __forceinline__ void wait_async0() {
  asm volatile("s_wait_asynccnt 0" ::: "memory");
}
__device__ __forceinline__ void wait_ds0() {
  asm volatile("s_wait_dscnt 0" ::: "memory");
}

// ---------------------------------------------------------------------------
// Generic bf16 GEMM: C[M,N] = A[M,K] x B[K,N], B supplied PRE-TRANSPOSED as
// Bt[N,K].  Block tile 128x128, 8 waves (4x2), wave tile 32x64, k-step 32.
// Tiles staged with GLOBAL_LOAD_ASYNC_TO_LDS_B128, double-buffered, one
// barrier per k-step.
// EPI: 0 = store bf16, 1 = relu + store bf16, 2 = add Res + store f32.
// ---------------------------------------------------------------------------
template<int EPI>
__global__ __launch_bounds__(256)
void gemm_bf16(const u16* __restrict__ A, const u16* __restrict__ Bt,
               const float* __restrict__ Res, float* __restrict__ OutF,
               u16* __restrict__ OutB, int M, int N, int K)
{
  (void)M;
  constexpr int LDT = 40;                     // padded LDS row stride (bf16)
  __shared__ u16 Al[2][128 * LDT];
  __shared__ u16 Bl[2][128 * LDT];
  const int tid  = threadIdx.x, lane = tid & 31, wid = tid >> 5;
  const int half = lane >> 4,   l16  = lane & 15;
  const int m0 = blockIdx.x * 128, n0 = blockIdx.y * 128;
  const int wm = (wid & 3) * 32,   wn = (wid >> 2) * 64;

  v8f acc[2][4];
#pragma unroll
  for (int i = 0; i < 2; ++i)
#pragma unroll
    for (int j = 0; j < 4; ++j)
      acc[i][j] = v8f{0.f,0.f,0.f,0.f,0.f,0.f,0.f,0.f};

  // 512 16-byte chunks per 128x32 tile; 2 chunks per thread
  const int r0 = tid >> 2,         ko0 = (tid & 3) * 8;
  const int r1 = (tid + 256) >> 2, ko1 = ((tid + 256) & 3) * 8;

  const u16* gA0 = A  + (size_t)(m0 + r0) * K + ko0;
  const u16* gA1 = A  + (size_t)(m0 + r1) * K + ko1;
  const u16* gB0 = Bt + (size_t)(n0 + r0) * K + ko0;
  const u16* gB1 = Bt + (size_t)(n0 + r1) * K + ko1;
  const unsigned lA0[2] = { (unsigned)(uintptr_t)&Al[0][r0 * LDT + ko0],
                            (unsigned)(uintptr_t)&Al[1][r0 * LDT + ko0] };
  const unsigned lA1[2] = { (unsigned)(uintptr_t)&Al[0][r1 * LDT + ko1],
                            (unsigned)(uintptr_t)&Al[1][r1 * LDT + ko1] };
  const unsigned lB0[2] = { (unsigned)(uintptr_t)&Bl[0][r0 * LDT + ko0],
                            (unsigned)(uintptr_t)&Bl[1][r0 * LDT + ko0] };
  const unsigned lB1[2] = { (unsigned)(uintptr_t)&Bl[0][r1 * LDT + ko1],
                            (unsigned)(uintptr_t)&Bl[1][r1 * LDT + ko1] };

  auto issue = [&](int ks, int buf) {
    async_cp16(lA0[buf], gA0 + ks);
    async_cp16(lA1[buf], gA1 + ks);
    async_cp16(lB0[buf], gB0 + ks);
    async_cp16(lB1[buf], gB1 + ks);
  };

  const int S = K >> 5;
  issue(0, 0);
  wait_async0();
  __syncthreads();

  for (int s = 0; s < S; ++s) {
    const int cur = s & 1;
    if (s + 1 < S) issue((s + 1) << 5, cur ^ 1);    // overlap with compute
    if (s + 2 < S) {
      __builtin_prefetch(gA0 + ((s + 2) << 5), 0, 1);
      __builtin_prefetch(gB0 + ((s + 2) << 5), 0, 1);
    }

    FragU AF[2], BF[4];
#pragma unroll
    for (int mt = 0; mt < 2; ++mt) {
      const u16* p = &Al[cur][(wm + mt * 16 + l16) * LDT];
      AF[mt].q[0] = *(const uint4*)(p + half * 8);
      AF[mt].q[1] = *(const uint4*)(p + 16 + half * 8);
    }
#pragma unroll
    for (int nt = 0; nt < 4; ++nt) {
      const u16* p = &Bl[cur][(wn + nt * 16 + l16) * LDT];
      BF[nt].q[0] = *(const uint4*)(p + half * 8);
      BF[nt].q[1] = *(const uint4*)(p + 16 + half * 8);
    }
#pragma unroll
    for (int mt = 0; mt < 2; ++mt)
#pragma unroll
      for (int nt = 0; nt < 4; ++nt)
        acc[mt][nt] = __builtin_amdgcn_wmma_f32_16x16x32_bf16(
            false, AF[mt].bf, false, BF[nt].bf, (short)0, acc[mt][nt],
            false, false);

    if (s + 1 < S) wait_async0();                   // own copies landed
    __syncthreads();                                // everyone's landed
  }

#pragma unroll
  for (int mt = 0; mt < 2; ++mt)
#pragma unroll
    for (int nt = 0; nt < 4; ++nt)
#pragma unroll
      for (int r = 0; r < 8; ++r) {
        const size_t gr = (size_t)(m0 + wm + mt * 16 + r + half * 8);
        const int    gc = n0 + wn + nt * 16 + l16;
        float v = acc[mt][nt][r];
        if constexpr (EPI == 2) {
          v += Res[gr * N + gc];
          OutF[gr * N + gc] = v;
        } else {
          if constexpr (EPI == 1) v = fmaxf(v, 0.f);
          OutB[gr * N + gc] = f2bf(v);
        }
      }
}

// ---------------------------------------------------------------------------
// Flash attention.  Grid (L/128, NH, B), 256 threads.  Wave = 16 query rows.
// Q,K in [B*L, 512] bf16 (col = h*64+d); Vt in [B, 512, L] bf16.
// Softmax runs in row-per-lane layout after an LDS transpose: one shfl_xor
// per reduction instead of a ds_bpermute butterfly.
// ---------------------------------------------------------------------------
__global__ __launch_bounds__(256)
void attn_fused(const u16* __restrict__ Qb, const u16* __restrict__ Kb,
                const u16* __restrict__ Vt, const float* __restrict__ mask,
                u16* __restrict__ Ob)
{
  constexpr int SLD = 36;                 // f32 LDS row stride (dwords-ish pad)
  constexpr int PLD = 40;                 // bf16 LDS row stride
  __shared__ float Sl[8][16 * SLD];
  __shared__ u16   Pl[8][16 * PLD];
  __shared__ float Aw[8][16];
  const int tid  = threadIdx.x, lane = tid & 31, wid = tid >> 5;
  const int half = lane >> 4,   l16  = lane & 15;
  const int h = blockIdx.y, b = blockIdx.z;
  const int q0 = blockIdx.x * 128 + wid * 16;
  const size_t rowbase = (size_t)b * L_ + q0;

  FragU QA[2];
  {
    const u16* qr = Qb + (rowbase + l16) * 512 + h * 64;
#pragma unroll
    for (int s = 0; s < 2; ++s) {
      QA[s].q[0] = *(const uint4*)(qr + s * 32 + half * 8);
      QA[s].q[1] = *(const uint4*)(qr + s * 32 + 16 + half * 8);
    }
  }

  float mprev = -1e30f, lsum = 0.f;       // stats for row = l16 (row layout)
  v8f O[4];
#pragma unroll
  for (int nt = 0; nt < 4; ++nt)
    O[nt] = v8f{0.f,0.f,0.f,0.f,0.f,0.f,0.f,0.f};

  for (int kv = 0; kv < L_; kv += 32) {
    // ---- S = Q K^T for 32 keys (two 16x16 C-layout tiles) ----
    v8f Sj[2];
#pragma unroll
    for (int j = 0; j < 2; ++j) {
      const u16* kr = Kb + ((size_t)b * L_ + kv + j * 16 + l16) * 512 + h * 64;
      v8f sc = v8f{0.f,0.f,0.f,0.f,0.f,0.f,0.f,0.f};
#pragma unroll
      for (int s = 0; s < 2; ++s) {
        FragU KB;
        KB.q[0] = *(const uint4*)(kr + s * 32 + half * 8);
        KB.q[1] = *(const uint4*)(kr + s * 32 + 16 + half * 8);
        sc = __builtin_amdgcn_wmma_f32_16x16x32_bf16(
            false, QA[s].bf, false, KB.bf, (short)0, sc, false, false);
      }
      Sj[j] = sc;
    }

    // ---- transpose S through LDS: C-layout -> row-per-lane ----
#pragma unroll
    for (int r = 0; r < 8; ++r) {
      Sl[wid][(r + half * 8) * SLD + l16]      = Sj[0][r];
      Sl[wid][(r + half * 8) * SLD + 16 + l16] = Sj[1][r];
    }
    wait_ds0();

    float sv[16];
    {
      const float* sp = &Sl[wid][l16 * SLD + half * 16];
#pragma unroll
      for (int c = 0; c < 4; ++c) {
        const float4 s4 = ((const float4*)sp)[c];
        sv[4 * c] = s4.x; sv[4 * c + 1] = s4.y;
        sv[4 * c + 2] = s4.z; sv[4 * c + 3] = s4.w;
      }
    }
    // scale + additive mask (vectorized 16B/lane reads of the mask stream)
    {
      const float* mp = mask + (rowbase + l16) * (size_t)L_ + kv + half * 16;
#pragma unroll
      for (int c = 0; c < 4; ++c) {
        const float4 m4 = ((const float4*)mp)[c];
        sv[4 * c]     = sv[4 * c]     * 0.125f + m4.x;
        sv[4 * c + 1] = sv[4 * c + 1] * 0.125f + m4.y;
        sv[4 * c + 2] = sv[4 * c + 2] * 0.125f + m4.z;
        sv[4 * c + 3] = sv[4 * c + 3] * 0.125f + m4.w;
      }
    }
    // ---- online softmax, one shfl per reduction ----
    float mx = sv[0];
#pragma unroll
    for (int i = 1; i < 16; ++i) mx = fmaxf(mx, sv[i]);
    mx = fmaxf(mx, __shfl_xor(mx, 16, 32));
    const float nm    = fmaxf(mprev, mx);
    const float alpha = __expf(mprev - nm);
    mprev = nm;

    float sum = 0.f;
    u16 pb[16];
#pragma unroll
    for (int i = 0; i < 16; ++i) {
      const float p = __expf(sv[i] - nm);
      sum += p;
      pb[i] = f2bf(p);
    }
    sum += __shfl_xor(sum, 16, 32);
    lsum = lsum * alpha + sum;

    // P (bf16, packed) -> LDS in A-fragment-friendly row layout
    {
      u16* pp = &Pl[wid][l16 * PLD + half * 16];
      unsigned w[8];
#pragma unroll
      for (int i = 0; i < 8; ++i)
        w[i] = (unsigned)pb[2 * i] | ((unsigned)pb[2 * i + 1] << 16);
      ((uint4*)pp)[0] = uint4{w[0], w[1], w[2], w[3]};
      *(uint4*)(pp + 8) = uint4{w[4], w[5], w[6], w[7]};
    }
    if (half == 0) Aw[wid][l16] = alpha;
    wait_ds0();

    // rescale O accumulators with per-row alpha (C-layout rows = r + 8*half)
    float aR[8];
#pragma unroll
    for (int r = 0; r < 8; ++r) aR[r] = Aw[wid][r + half * 8];
#pragma unroll
    for (int nt = 0; nt < 4; ++nt)
#pragma unroll
      for (int r = 0; r < 8; ++r) O[nt][r] *= aR[r];

    // ---- O += P V ----
    FragU PA;
    {
      const u16* p = &Pl[wid][l16 * PLD];
      PA.q[0] = *(const uint4*)(p + half * 8);
      PA.q[1] = *(const uint4*)(p + 16 + half * 8);
    }
#pragma unroll
    for (int nt = 0; nt < 4; ++nt) {
      const u16* vr = Vt + ((size_t)b * 512 + h * 64 + nt * 16 + l16) * L_
                        + kv + half * 8;
      FragU VB;
      VB.q[0] = *(const uint4*)vr;
      VB.q[1] = *(const uint4*)(vr + 16);
      O[nt] = __builtin_amdgcn_wmma_f32_16x16x32_bf16(
          false, PA.bf, false, VB.bf, (short)0, O[nt], false, false);
    }
  }

  // ---- epilogue: divide by per-row l, store bf16 ----
  if (half == 0) Aw[wid][l16] = lsum;
  wait_ds0();
  float lR[8];
#pragma unroll
  for (int r = 0; r < 8; ++r) lR[r] = 1.f / Aw[wid][r + half * 8];
#pragma unroll
  for (int nt = 0; nt < 4; ++nt)
#pragma unroll
    for (int r = 0; r < 8; ++r) {
      const float v = O[nt][r] * lR[r];
      Ob[(rowbase + r + half * 8) * 512 + h * 64 + nt * 16 + l16] = f2bf(v);
    }
}

// ---------------------------------------------------------------------------
// LayerNorm over 512-wide rows; one wave per row.
// ---------------------------------------------------------------------------
__global__ __launch_bounds__(256)
void layernorm_512(const float* __restrict__ X, float* __restrict__ Yf,
                   u16* __restrict__ Yb)
{
  const int row  = blockIdx.x * 8 + (threadIdx.x >> 5);
  const int lane = threadIdx.x & 31;
  const float* x = X + (size_t)row * 512;
  float v[16], s = 0.f, s2 = 0.f;
#pragma unroll
  for (int i = 0; i < 16; ++i) {
    v[i] = x[lane + 32 * i];
    s += v[i]; s2 += v[i] * v[i];
  }
#pragma unroll
  for (int d = 1; d < 32; d <<= 1) {
    s  += __shfl_xor(s,  d, 32);
    s2 += __shfl_xor(s2, d, 32);
  }
  const float mean = s * (1.f / 512.f);
  const float var  = s2 * (1.f / 512.f) - mean * mean;
  const float rstd = rsqrtf(var + 1e-5f);
#pragma unroll
  for (int i = 0; i < 16; ++i) {
    const float y = (v[i] - mean) * rstd;
    Yf[(size_t)row * 512 + lane + 32 * i] = y;
    if (Yb) Yb[(size_t)row * 512 + lane + 32 * i] = f2bf(y);
  }
}

// ---------------------------------------------------------------------------
// Conversion / transpose helpers (run once per launch; bandwidth-trivial).
// ---------------------------------------------------------------------------
__global__ void cvt_f32_bf16(const float* __restrict__ X, u16* __restrict__ Y,
                             size_t n)
{
  for (size_t i = blockIdx.x * (size_t)blockDim.x + threadIdx.x; i < n;
       i += (size_t)gridDim.x * blockDim.x)
    Y[i] = f2bf(X[i]);
}

__global__ void cvt_f32_bf16_T(const float* __restrict__ W,
                               u16* __restrict__ Wt, int Kd, int Nd)
{
  const size_t n = (size_t)Kd * Nd;
  for (size_t i = blockIdx.x * (size_t)blockDim.x + threadIdx.x; i < n;
       i += (size_t)gridDim.x * blockDim.x) {
    const int k = (int)(i / Nd), c = (int)(i % Nd);
    Wt[(size_t)c * Kd + k] = f2bf(W[i]);
  }
}

__global__ void transpose_v_bf16(const u16* __restrict__ V,
                                 u16* __restrict__ Vt)
{
  const size_t n = (size_t)B_ * L_ * 512;
  for (size_t i = blockIdx.x * (size_t)blockDim.x + threadIdx.x; i < n;
       i += (size_t)gridDim.x * blockDim.x) {
    const size_t row = i >> 9;
    const int col = (int)(i & 511);
    const int b = (int)(row / L_), l = (int)(row % L_);
    Vt[((size_t)b * 512 + col) * L_ + l] = V[i];
  }
}

// ---------------------------------------------------------------------------
extern "C" void kernel_launch(void* const* d_in, const int* in_sizes, int n_in,
                              void* d_out, int out_size, void* d_ws,
                              size_t ws_size, hipStream_t stream)
{
  (void)in_sizes; (void)n_in; (void)out_size; (void)ws_size;
  const float* query = (const float*)d_in[0];
  const float* keyx  = (const float*)d_in[1];
  const float* value = (const float*)d_in[2];
  const float* mask  = (const float*)d_in[3];
  const float* WQ = (const float*)d_in[4];
  const float* WK = (const float*)d_in[5];
  const float* WV = (const float*)d_in[6];
  const float* WO = (const float*)d_in[7];
  const float* W1 = (const float*)d_in[8];
  const float* W2 = (const float*)d_in[9];
  float* outp = (float*)d_out;

  char* ws = (char*)d_ws;
  size_t off = 0;
  auto take = [&](size_t bytes) -> void* {
    void* p = ws + off;
    off = (off + bytes + 255) & ~(size_t)255;
    return p;
  };
  const size_t R = (size_t)B_ * L_;   // 8192 rows
  u16* Xq  = (u16*)take(R * 512 * 2);
  u16* Xk  = (u16*)take(R * 512 * 2);
  u16* Xv  = (u16*)take(R * 512 * 2);
  u16* WQt = (u16*)take((size_t)512 * 512 * 2);
  u16* WKt = (u16*)take((size_t)512 * 512 * 2);
  u16* WVt = (u16*)take((size_t)512 * 512 * 2);
  u16* WOt = (u16*)take((size_t)512 * 512 * 2);
  u16* W1t = (u16*)take((size_t)512 * 2048 * 2);
  u16* W2t = (u16*)take((size_t)512 * 2048 * 2);
  u16* Qb  = (u16*)take(R * 512 * 2);
  u16* Kb  = (u16*)take(R * 512 * 2);
  u16* Vb  = (u16*)take(R * 512 * 2);
  u16* Vt  = (u16*)take(R * 512 * 2);
  u16* Ab  = (u16*)take(R * 512 * 2);
  float* Proj = (float*)take(R * 512 * 4);
  float* LN1f = (float*)take(R * 512 * 4);
  u16*   LN1b = (u16*)take(R * 512 * 2);
  u16*   Hb   = (u16*)take(R * 2048 * 2);
  float* FFf  = (float*)take(R * 512 * 4);

  const dim3 thr(256);
  cvt_f32_bf16<<<2048, thr, 0, stream>>>(query, Xq, R * 512);
  cvt_f32_bf16<<<2048, thr, 0, stream>>>(keyx,  Xk, R * 512);
  cvt_f32_bf16<<<2048, thr, 0, stream>>>(value, Xv, R * 512);
  cvt_f32_bf16_T<<<1024, thr, 0, stream>>>(WQ, WQt, 512, 512);
  cvt_f32_bf16_T<<<1024, thr, 0, stream>>>(WK, WKt, 512, 512);
  cvt_f32_bf16_T<<<1024, thr, 0, stream>>>(WV, WVt, 512, 512);
  cvt_f32_bf16_T<<<1024, thr, 0, stream>>>(WO, WOt, 512, 512);
  cvt_f32_bf16_T<<<2048, thr, 0, stream>>>(W1, W1t, 512, 2048);
  cvt_f32_bf16_T<<<2048, thr, 0, stream>>>(W2, W2t, 2048, 512);

  // Q/K/V projections -> bf16 [B*L, 512] (col = h*64+d)
  gemm_bf16<0><<<dim3(64, 4), thr, 0, stream>>>(Xq, WQt, nullptr, nullptr, Qb,
                                                8192, 512, 512);
  gemm_bf16<0><<<dim3(64, 4), thr, 0, stream>>>(Xk, WKt, nullptr, nullptr, Kb,
                                                8192, 512, 512);
  gemm_bf16<0><<<dim3(64, 4), thr, 0, stream>>>(Xv, WVt, nullptr, nullptr, Vb,
                                                8192, 512, 512);
  transpose_v_bf16<<<2048, thr, 0, stream>>>(Vb, Vt);

  attn_fused<<<dim3(16, NH, B_), thr, 0, stream>>>(Qb, Kb, Vt, mask, Ab);

  // attn @ WO + query residual -> f32, then LN1 (f32 + bf16 copies)
  gemm_bf16<2><<<dim3(64, 4), thr, 0, stream>>>(Ab, WOt, query, Proj, nullptr,
                                                8192, 512, 512);
  layernorm_512<<<1024, thr, 0, stream>>>(Proj, LN1f, LN1b);

  // FFN: relu(LN1 @ W1) @ W2 + LN1 residual, then LN2 -> d_out
  gemm_bf16<1><<<dim3(64, 16), thr, 0, stream>>>(LN1b, W1t, nullptr, nullptr,
                                                 Hb, 8192, 2048, 512);
  gemm_bf16<2><<<dim3(64, 4), thr, 0, stream>>>(Hb, W2t, LN1f, FFf, nullptr,
                                                8192, 512, 2048);
  layernorm_512<<<1024, thr, 0, stream>>>(FFf, outp, nullptr);
}
